// LSTMPCell_57380763074830
// MI455X (gfx1250) — matrix operand order repaired and checked by reading, not compiled, not run on hardware
//
#include <hip/hip_runtime.h>

// ---------------------------------------------------------------------------
// LSTM-with-projection for MI455X (gfx1250, wave32, WMMA).
//   T=512 B=32 D=512 H=512 C=2048
// Recurrent step: gates(32x8192) = [x_t ; h](32x1024) @ [Wi|Wh]^T, K fused.
// bf16 operands (converted once into workspace, weights stay hot in 192MB L2),
// fp32 accumulation via v_wmma_f32_16x16x32_bf16.
// K is split across waves (partial-sum reduction through LDS) so the
// latency-bound recurrent step runs with 1024 / 512 waves instead of 256 / 64.
// ---------------------------------------------------------------------------

typedef __bf16 bf16;
typedef __attribute__((ext_vector_type(8)))  bf16  bf16x8;
typedef __attribute__((ext_vector_type(16))) bf16  v16bf;
typedef __attribute__((ext_vector_type(8)))  float v8f;

#define T_STEPS 512
#define BATCH   32
#define DIN     512
#define HID     512
#define CELL    2048
#define GATES   (4 * CELL)   // 8192
#define KTOT    (DIN + HID)  // 1024

// round-to-nearest-even fp32 -> bf16
__device__ __forceinline__ bf16 f2bf(float f) {
  unsigned u = __builtin_bit_cast(unsigned, f);
  u += 0x7FFFu + ((u >> 16) & 1u);
  unsigned short s = (unsigned short)(u >> 16);
  return __builtin_bit_cast(bf16, s);
}

__device__ __forceinline__ float sigm(float x) { return 1.0f / (1.0f + __expf(-x)); }

// Build a 16-element bf16 fragment from two 16B contiguous chunks.
__device__ __forceinline__ v16bf ld_frag(const bf16* p0, const bf16* p1) {
  bf16x8 lo = *(const bf16x8*)p0;
  bf16x8 hi = *(const bf16x8*)p1;
  return __builtin_shufflevector(lo, hi, 0, 1, 2, 3, 4, 5, 6, 7,
                                 8, 9, 10, 11, 12, 13, 14, 15);
}

__device__ __forceinline__ v8f wmma_bf16(v16bf a, v16bf b, v8f c) {
  return __builtin_amdgcn_wmma_f32_16x16x32_bf16(
      /*neg_a=*/false, a, /*neg_b=*/false, b,
      /*c_mod=*/(short)0, c, /*reuse_a=*/false, /*reuse_b=*/false);
}

// ---------------------------------------------------------------------------
// One-time prep: pack [Wi|Wh] rows to bf16, cast Wp / input to bf16, zero state
// ---------------------------------------------------------------------------
__global__ void pack_w(const float* __restrict__ Wi, const float* __restrict__ Wh,
                       bf16* __restrict__ W16) {
  size_t i = (size_t)blockIdx.x * blockDim.x + threadIdx.x;
  if (i >= (size_t)GATES * KTOT) return;
  int k = (int)(i & (KTOT - 1));
  size_t g = i >> 10;
  float v = (k < DIN) ? Wi[g * DIN + k] : Wh[g * HID + (k - DIN)];
  W16[i] = f2bf(v);
}

__global__ void pack_cast(const float* __restrict__ src, bf16* __restrict__ dst,
                          size_t n) {
  size_t i = (size_t)blockIdx.x * blockDim.x + threadIdx.x;
  if (i < n) dst[i] = f2bf(src[i]);
}

__global__ void init_state(float* __restrict__ Hf32, bf16* __restrict__ H16,
                           float* __restrict__ Cst) {
  int i = blockIdx.x * blockDim.x + threadIdx.x;
  if (i < BATCH * HID) { Hf32[i] = 0.0f; H16[i] = f2bf(0.0f); }
  if (i < BATCH * CELL) Cst[i] = 0.0f;
}

// ---------------------------------------------------------------------------
// Gate + cell kernel (per timestep).
// grid = 128 blocks (cell-column tiles of 16) x 256 threads (8 waves).
// wave = (mtile in 0..1) x (kq in 0..3): each wave accumulates a K=256 slice
// (kq 0..1 -> x_t part, kq 2..3 -> h part) of the i/f/g/o tiles for the SAME
// 16 cell columns. Partials reduced through LDS; kq==0 waves run the fused
// sigmoid/tanh cell update.
// ---------------------------------------------------------------------------
__global__ void __launch_bounds__(256)
lstm_gate_cell(const bf16* __restrict__ X16, const bf16* __restrict__ H16,
               const bf16* __restrict__ W16, const float* __restrict__ bias,
               const int* __restrict__ lens, float* __restrict__ Cst,
               bf16* __restrict__ Abuf, int t) {
  const int lane  = threadIdx.x & 31;
  const int wave  = threadIdx.x >> 5;  // 0..7
  const int mtile = wave & 1;          // batch rows [mtile*16, mtile*16+16)
  const int kq    = wave >> 1;         // K slice [kq*256, kq*256+256)
  const int jt    = blockIdx.x;        // cell-column tile
  const int half  = lane >> 4;         // 0 / 1 (half-wave)
  const int col   = lane & 15;

  // A fragment source row: batch row m = mtile*16 + col.
  // 16-bit A layout: lane half h holds K = {8h..8h+7} and {16+8h..16+8h+7}.
  const int m = mtile * 16 + col;
  const bf16* arow;
  if (kq < 2)
    arow = X16 + ((size_t)t * BATCH + m) * DIN + kq * 256 + half * 8;
  else
    arow = H16 + (size_t)m * HID + (kq - 2) * 256 + half * 8;

  // B fragment source rows: g = q*CELL + jt*16 + col, same K layout as A:
  // lane half h holds K = {8h..8h+7} and {16+8h..16+8h+7}.
  const bf16* brow[4];
#pragma unroll
  for (int q = 0; q < 4; ++q)
    brow[q] = W16 + (size_t)(q * CELL + jt * 16 + col) * KTOT + kq * 256 +
              half * 8;

  v8f zero = {0.f, 0.f, 0.f, 0.f, 0.f, 0.f, 0.f, 0.f};
  v8f acc[4] = {zero, zero, zero, zero};

#pragma unroll
  for (int kb = 0; kb < 256; kb += 32) {
    v16bf a = ld_frag(arow + kb, arow + kb + 16);
#pragma unroll
    for (int q = 0; q < 4; ++q) {
      v16bf b = ld_frag(brow[q] + kb, brow[q] + kb + 16);
      acc[q] = wmma_bf16(a, b, acc[q]);
    }
  }

  // Cross-wave K reduction through LDS (kq 1..3 contribute, kq 0 reduces).
  __shared__ float red[3][2][4][8][32];   // 24 KB
  if (kq > 0) {
#pragma unroll
    for (int q = 0; q < 4; ++q)
#pragma unroll
      for (int r = 0; r < 8; ++r)
        red[kq - 1][mtile][q][r][lane] = acc[q][r];
  }
  __syncthreads();
  if (kq != 0) return;

#pragma unroll
  for (int q = 0; q < 4; ++q)
#pragma unroll
    for (int r = 0; r < 8; ++r)
      acc[q][r] += red[0][mtile][q][r][lane] + red[1][mtile][q][r][lane] +
                   red[2][mtile][q][r][lane];

  // Epilogue: D layout -> VGPR r holds (M = r + 8*half, N = col).
  const int j = jt * 16 + col;
  const float bi  = bias[j];
  const float bff = bias[CELL + j];
  const float bg  = bias[2 * CELL + j];
  const float bo  = bias[3 * CELL + j];
#pragma unroll
  for (int r = 0; r < 8; ++r) {
    int mg = mtile * 16 + half * 8 + r;
    float iv = sigm(acc[0][r] + bi);
    float fv = sigm(acc[1][r] + bff);
    float gv = tanhf(acc[2][r] + bg);
    float ov = sigm(acc[3][r] + bo);
    size_t cidx = (size_t)mg * CELL + j;
    float cold = Cst[cidx];
    float cnew = fv * cold + iv * gv;
    bool act = lens[mg] > t;
    Cst[cidx] = act ? cnew : cold;
    Abuf[cidx] = f2bf(ov * tanhf(cnew));   // masked rows discarded downstream
  }
}

// ---------------------------------------------------------------------------
// Projection kernel: h_new(32x512) = Abuf(32x2048) @ Wp^T, masked state update.
// grid = 32 blocks (h-column tiles) x 512 threads (16 waves).
// wave = (mtile in 0..1) x (kq in 0..7, K=256 slices); LDS reduction.
// ---------------------------------------------------------------------------
__global__ void __launch_bounds__(512)
lstm_proj(const bf16* __restrict__ Abuf, const bf16* __restrict__ Wp16,
          const int* __restrict__ lens, float* __restrict__ Hf32,
          bf16* __restrict__ H16, float* __restrict__ out, int t) {
  const int lane  = threadIdx.x & 31;
  const int wave  = threadIdx.x >> 5;  // 0..15
  const int mtile = wave & 1;
  const int kq    = wave >> 1;         // 0..7
  const int nt    = blockIdx.x;        // h-column tile
  const int half  = lane >> 4;
  const int col   = lane & 15;

  const bf16* arow =
      Abuf + (size_t)(mtile * 16 + col) * CELL + kq * 256 + half * 8;
  const bf16* brow =
      Wp16 + (size_t)(nt * 16 + col) * CELL + kq * 256 + half * 8;

  v8f acc = {0.f, 0.f, 0.f, 0.f, 0.f, 0.f, 0.f, 0.f};
#pragma unroll
  for (int kb = 0; kb < 256; kb += 32) {
    v16bf a = ld_frag(arow + kb, arow + kb + 16);
    v16bf b = ld_frag(brow + kb, brow + kb + 16);
    acc = wmma_bf16(a, b, acc);
  }

  __shared__ float red[7][2][8][32];   // 14 KB
  if (kq > 0) {
#pragma unroll
    for (int r = 0; r < 8; ++r)
      red[kq - 1][mtile][r][lane] = acc[r];
  }
  __syncthreads();
  if (kq != 0) return;

#pragma unroll
  for (int r = 0; r < 8; ++r) {
    float s = acc[r];
#pragma unroll
    for (int p = 0; p < 7; ++p) s += red[p][mtile][r][lane];
    acc[r] = s;
  }

  const int hc = nt * 16 + col;
#pragma unroll
  for (int r = 0; r < 8; ++r) {
    int mg = mtile * 16 + half * 8 + r;
    float hnew = acc[r];
    bool act = lens[mg] > t;
    size_t hidx = (size_t)mg * HID + hc;
    float hold = Hf32[hidx];
    float h2 = act ? hnew : hold;
    Hf32[hidx] = h2;
    H16[hidx] = f2bf(h2);
    out[((size_t)t * BATCH + mg) * HID + hc] = act ? hnew : 0.0f;
  }
}

// Copy final (hT, cT) into the output tail.
__global__ void finalize(const float* __restrict__ Hf32,
                         const float* __restrict__ Cst,
                         float* __restrict__ tail) {
  int i = blockIdx.x * blockDim.x + threadIdx.x;
  if (i < BATCH * HID) tail[i] = Hf32[i];
  if (i < BATCH * CELL) tail[BATCH * HID + i] = Cst[i];
}

// ---------------------------------------------------------------------------
extern "C" void kernel_launch(void* const* d_in, const int* in_sizes, int n_in,
                              void* d_out, int out_size, void* d_ws, size_t ws_size,
                              hipStream_t stream) {
  (void)in_sizes; (void)n_in; (void)out_size; (void)ws_size;
  const float* input = (const float*)d_in[0];
  const int*   lens  = (const int*)d_in[1];
  const float* Wi    = (const float*)d_in[2];
  const float* Wh    = (const float*)d_in[3];
  const float* bias  = (const float*)d_in[4];
  const float* Wp    = (const float*)d_in[5];
  float* out = (float*)d_out;

  // Workspace carve-out (256B aligned), total ~36 MB.
  char* ws = (char*)d_ws;
  size_t off = 0;
  auto carve = [&](size_t bytes) -> void* {
    void* p = ws + off;
    off += (bytes + 255) & ~(size_t)255;
    return p;
  };
  bf16*  W16  = (bf16*) carve((size_t)GATES * KTOT * sizeof(bf16));     // 16 MB
  bf16*  Wp16 = (bf16*) carve((size_t)HID * CELL * sizeof(bf16));       //  2 MB
  bf16*  X16  = (bf16*) carve((size_t)T_STEPS * BATCH * DIN * 2);       // 16 MB
  float* Hf32 = (float*)carve((size_t)BATCH * HID * sizeof(float));
  bf16*  H16  = (bf16*) carve((size_t)BATCH * HID * sizeof(bf16));
  float* Cst  = (float*)carve((size_t)BATCH * CELL * sizeof(float));
  bf16*  Abuf = (bf16*) carve((size_t)BATCH * CELL * sizeof(bf16));

  // Phase 0: pack weights/input to bf16, zero recurrent state.
  size_t nW = (size_t)GATES * KTOT;
  pack_w<<<(unsigned)((nW + 255) / 256), 256, 0, stream>>>(Wi, Wh, W16);
  size_t nWp = (size_t)HID * CELL;
  pack_cast<<<(unsigned)((nWp + 255) / 256), 256, 0, stream>>>(Wp, Wp16, nWp);
  size_t nX = (size_t)T_STEPS * BATCH * DIN;
  pack_cast<<<(unsigned)((nX + 255) / 256), 256, 0, stream>>>(input, X16, nX);
  init_state<<<(BATCH * CELL + 255) / 256, 256, 0, stream>>>(Hf32, H16, Cst);

  // Recurrence: sequential launches provide the grid-wide sync per step.
  for (int t = 0; t < T_STEPS; ++t) {
    lstm_gate_cell<<<CELL / 16, 256, 0, stream>>>(X16, H16, W16, bias, lens,
                                                  Cst, Abuf, t);
    lstm_proj<<<HID / 16, 512, 0, stream>>>(Abuf, Wp16, lens, Hf32, H16, out, t);
  }
  finalize<<<(BATCH * CELL + 255) / 256, 256, 0, stream>>>(
      Hf32, Cst, out + (size_t)T_STEPS * BATCH * HID);
}
